// CausalSelfAttention_27693949125402
// MI455X (gfx1250) — matrix operand order repaired and checked by reading, not compiled, not run on hardware
//
#include <hip/hip_runtime.h>
#include <hip/hip_bf16.h>
#include <stdint.h>

#define B_   4
#define T_   2048
#define HID_ 768
#define H_   12
#define D_   64

typedef __attribute__((ext_vector_type(16))) __bf16 v16bf;
typedef __attribute__((ext_vector_type(8)))  float  v8f;

union Frag16 {
    v16bf v;
    unsigned short u[16];
};

// f32 -> bf16 round-to-nearest-even (bits), scalar
__device__ __forceinline__ unsigned short f2bf(float f) {
    union { float f; unsigned u; } a; a.f = f;
    unsigned r = a.u + 0x7FFFu + ((a.u >> 16) & 1u);
    return (unsigned short)(r >> 16);
}

// pack two f32 -> {bf16(x) lo, bf16(y) hi} in one dword via v_perm_b32
__device__ __forceinline__ unsigned pk2bf(float x, float y) {
    union { float f; unsigned u; } a, b; a.f = x; b.f = y;
    unsigned ra = a.u + 0x7FFFu + ((a.u >> 16) & 1u);
    unsigned rb = b.u + 0x7FFFu + ((b.u >> 16) & 1u);
    return __builtin_amdgcn_perm(rb, ra, 0x07060302);  // {rb.hi16, ra.hi16}
}

// Split workgroup barrier draining ONLY the DS counter (LDS producer/consumer
// visibility). Unlike __syncthreads(), this does NOT join in-flight global
// loads / async copies, so pipelined VMEM stays in flight across the barrier.
__device__ __forceinline__ void wg_barrier_lds() {
    asm volatile("s_wait_dscnt 0\n\t"
                 "s_barrier_signal -1\n\t"
                 "s_barrier_wait -1" ::: "memory");
}

// ASYNCcnt waits (async loads complete in order)
__device__ __forceinline__ void wait_async_le8() {
    asm volatile("s_wait_asynccnt 8" ::: "memory");
}
__device__ __forceinline__ void wait_async_0() {
    asm volatile("s_wait_asynccnt 0" ::: "memory");
}

// Async DMA of a 64x64 bf16 tile (row stride D_=64) straight into padded LDS:
// 4x GLOBAL_LOAD_ASYNC_TO_LDS_B128 per lane, no VGPR landing, ASYNCcnt-tracked.
// LDS operand = low 32 bits of the generic pointer (LDS aperture offset).
__device__ __forceinline__ void async_tile64(unsigned short (*dst)[72],
                                             const unsigned short* __restrict__ src,
                                             int tid) {
    int r  = tid >> 3;          // 0..15
    int c8 = (tid & 7) * 8;     // 0..56
    #pragma unroll
    for (int p = 0; p < 4; ++p) {
        int row = p * 16 + r;
        unsigned lds = (unsigned)(uintptr_t)&dst[row][c8];
        const void* g = (const void*)(src + (size_t)row * D_ + c8);
        asm volatile("global_load_async_to_lds_b128 %0, %1, off"
                     :: "v"(lds), "v"(g) : "memory");
    }
}

// ---------------------------------------------------------------------------
// Kernel 1: fused QKV projection, software-pipelined.
//   y = x @ W^T + b  (f32 in, bf16 [B,H,T,D] out; Q pre-scaled by 1/sqrt(D))
// grid = (B*T/128, H, 3{q,k,v}), block = 256 (8 waves); each wave owns a
// 16x64 output tile, K=768 swept in 32-wide bf16 WMMA steps. Per iteration:
// commit staged regs -> LDS, dscnt-barrier, issue next chunk's global loads,
// WMMA from LDS; the loadcnt join lands at the NEXT commit (first use), so
// HBM loads overlap the WMMA burst and both barriers.
// ---------------------------------------------------------------------------
__global__ __launch_bounds__(256)
void qkv_gemm_kernel(const float* __restrict__ X,
                     const float* __restrict__ Wq, const float* __restrict__ Wk,
                     const float* __restrict__ Wv,
                     const float* __restrict__ bq, const float* __restrict__ bk,
                     const float* __restrict__ bv,
                     unsigned short* __restrict__ Qo,
                     unsigned short* __restrict__ Ko,
                     unsigned short* __restrict__ Vo) {
    // padded to 36 halfwords (72B = 18 banks): conflict-free fragment reads
    __shared__ __align__(16) unsigned short xs[128][36];
    __shared__ __align__(16) unsigned short ws[64][36];

    const int tid   = threadIdx.x;
    const int wave  = tid >> 5;       // 0..7
    const int lane  = tid & 31;
    const int ln    = lane & 15;
    const int hi    = lane >> 4;      // half-wave select
    const int mtile = blockIdx.x;     // 0..63  (128 rows each)
    const int h     = blockIdx.y;     // head 0..11
    const int which = blockIdx.z;     // 0=q 1=k 2=v

    const float* W  = (which == 0) ? Wq : (which == 1) ? Wk : Wv;
    const float* bb = (which == 0) ? bq : (which == 1) ? bk : bv;
    unsigned short* out = (which == 0) ? Qo : (which == 1) ? Ko : Vo;
    const float oscale = (which == 0) ? 0.125f : 1.0f;  // fold 1/sqrt(D) into Q

    const v8f zero8 = {0.f, 0.f, 0.f, 0.f, 0.f, 0.f, 0.f, 0.f};
    v8f acc[4] = {zero8, zero8, zero8, zero8};

    const int r0 = tid >> 3;          // 0..31
    const int c4 = (tid & 7) * 4;     // 0..28

    // preload chunk 0 into registers
    float4 xr[4], wr[2];
    #pragma unroll
    for (int p = 0; p < 4; ++p)
        xr[p] = *(const float4*)&X[(size_t)(mtile * 128 + p * 32 + r0) * HID_ + c4];
    #pragma unroll
    for (int p = 0; p < 2; ++p)
        wr[p] = *(const float4*)&W[(size_t)(h * 64 + p * 32 + r0) * HID_ + c4];

    for (int kc = 0; kc < HID_ / 32; ++kc) {
        // commit staged registers to LDS (packed bf16 pairs, 32-bit stores)
        #pragma unroll
        for (int p = 0; p < 4; ++p) {
            int row = p * 32 + r0;
            *(unsigned*)&xs[row][c4]     = pk2bf(xr[p].x, xr[p].y);
            *(unsigned*)&xs[row][c4 + 2] = pk2bf(xr[p].z, xr[p].w);
        }
        #pragma unroll
        for (int p = 0; p < 2; ++p) {
            int row = p * 32 + r0;
            *(unsigned*)&ws[row][c4]     = pk2bf(wr[p].x, wr[p].y);
            *(unsigned*)&ws[row][c4 + 2] = pk2bf(wr[p].z, wr[p].w);
        }
        wg_barrier_lds();

        // issue next chunk's global loads; they stay in flight under the WMMAs
        // below AND across the trailing barrier (dscnt-only)
        if (kc + 1 < HID_ / 32) {
            int ko = (kc + 1) * 32 + c4;
            #pragma unroll
            for (int p = 0; p < 4; ++p)
                xr[p] = *(const float4*)&X[(size_t)(mtile * 128 + p * 32 + r0) * HID_ + ko];
            #pragma unroll
            for (int p = 0; p < 2; ++p)
                wr[p] = *(const float4*)&W[(size_t)(h * 64 + p * 32 + r0) * HID_ + ko];
        }

        // A fragment: 16x32, lane = row, K pattern per ISA layout
        Frag16 a;
        #pragma unroll
        for (int e = 0; e < 16; ++e) {
            int k = (e >> 3) * 16 + hi * 8 + (e & 7);
            a.u[e] = xs[wave * 16 + ln][k];
        }
        #pragma unroll
        for (int s = 0; s < 4; ++s) {
            // B fragment: 32x16, lane = column (output n), K split by half-wave
            Frag16 bfr;
            #pragma unroll
            for (int e = 0; e < 16; ++e)
                bfr.u[e] = ws[s * 16 + ln][(e >> 3) * 16 + hi * 8 + (e & 7)];
            acc[s] = __builtin_amdgcn_wmma_f32_16x16x32_bf16(
                false, a.v, false, bfr.v, (short)0, acc[s], false, false);
        }
        wg_barrier_lds();   // all waves done reading before next commit
    }

    // epilogue: + bias, optional Q scale, cast bf16, scatter into [B,H,T,D]
    #pragma unroll
    for (int s = 0; s < 4; ++s) {
        int n = s * 16 + ln;                 // 0..63 within head
        float bias = bb[h * 64 + n];
        #pragma unroll
        for (int r = 0; r < 8; ++r) {
            int grow = mtile * 128 + wave * 16 + r + hi * 8;   // global token row
            int bidx = grow / T_, t = grow % T_;
            float val = (acc[s][r] + bias) * oscale;
            out[(((size_t)bidx * H_ + h) * T_ + t) * D_ + n] = f2bf(val);
        }
    }
}

// ---------------------------------------------------------------------------
// Kernel 2: flash-attention (online softmax), causal + additive mask.
// grid = (T/64, B*H), block = 128 (4 waves). Block owns 64 queries; each wave
// owns 16. Single pass over key tiles 0..qt. K/V tiles are double-buffered in
// LDS and filled by GLOBAL_LOAD_ASYNC_TO_LDS_B128 (ASYNCcnt): tile kt+1's DMA
// runs under tile kt's WMMAs. S and O via bf16 WMMA; softmax state in f32
// registers; T x T scores never touch memory. Q arrives pre-scaled.
// ---------------------------------------------------------------------------
__global__ __launch_bounds__(128)
void attn_kernel(const unsigned short* __restrict__ Q,
                 const unsigned short* __restrict__ K,
                 const unsigned short* __restrict__ V,
                 const float* __restrict__ amask,   // [B, T] additive (over keys)
                 float* __restrict__ out) {         // [B, T, HID]
    __shared__ __align__(16) unsigned short ks[2][64][72];
    __shared__ __align__(16) unsigned short vs[2][64][72];
    __shared__ __align__(16) unsigned short ps[64][72];  // Q staging, then P

    const int tid  = threadIdx.x;
    const int wv   = tid >> 5;        // wave 0..3
    const int lane = tid & 31;
    const int ln   = lane & 15;
    const int hi   = lane >> 4;
    const int qt   = blockIdx.x;      // query tile 0..31
    const int bh   = blockIdx.y;      // 0..47
    const int b    = bh / H_;
    const int h    = bh % H_;
    const size_t base = (size_t)bh * T_ * D_;
    const int q0 = qt * 64;

    // async-stage Q, then tile 0 of K/V behind it (4 + 8 ops in flight;
    // completion is in-order, so asynccnt<=8 certifies Q is in LDS)
    async_tile64(ps,    Q + base + (size_t)q0 * D_, tid);
    async_tile64(ks[0], K + base, tid);
    async_tile64(vs[0], V + base, tid);
    wait_async_le8();
    wg_barrier_lds();

    // pull the two 16x32 Q A-fragments into registers (wave-private rows)
    Frag16 qa[2];
    #pragma unroll
    for (int kc = 0; kc < 2; ++kc)
        #pragma unroll
        for (int e = 0; e < 16; ++e) {
            int k = kc * 32 + (e >> 3) * 16 + hi * 8 + (e & 7);
            qa[kc].u[e] = ps[wv * 16 + ln][k];
        }

    const v8f zero8 = {0.f, 0.f, 0.f, 0.f, 0.f, 0.f, 0.f, 0.f};
    v8f o[4] = {zero8, zero8, zero8, zero8};
    float m[8], l[8];
    #pragma unroll
    for (int r = 0; r < 8; ++r) { m[r] = -1e30f; l[r] = 0.f; }

    for (int kt = 0; kt <= qt; ++kt) {
        const int cur = kt & 1;
        // (a) all waves done reading buffer cur^1 (previous tile) -> reusable
        wg_barrier_lds();
        if (kt < qt) {
            async_tile64(ks[cur ^ 1], K + base + (size_t)(kt + 1) * 64 * D_, tid);
            async_tile64(vs[cur ^ 1], V + base + (size_t)(kt + 1) * 64 * D_, tid);
            wait_async_le8();   // 16 in flight; oldest 8 (= tile kt) complete
        } else {
            wait_async_0();     // last tile: drain
        }
        // (b) tile kt visible to all waves; tile kt+1 DMA keeps running
        wg_barrier_lds();

        // S = Q K^T  (16 queries x 64 keys per wave; Q pre-scaled)
        v8f sacc[4] = {zero8, zero8, zero8, zero8};
        #pragma unroll
        for (int kc = 0; kc < 2; ++kc)
            #pragma unroll
            for (int s = 0; s < 4; ++s) {
                Frag16 kb;   // B[k=d][n=key] = K[key][d]
                #pragma unroll
                for (int e = 0; e < 16; ++e)
                    kb.u[e] = ks[cur][s * 16 + ln][kc * 32 + (e >> 3) * 16 + hi * 8 + (e & 7)];
                sacc[s] = __builtin_amdgcn_wmma_f32_16x16x32_bf16(
                    false, qa[kc].v, false, kb.v, (short)0, sacc[s], false, false);
            }

        // causal + additive mask
        float sc[4][8];
        #pragma unroll
        for (int s = 0; s < 4; ++s) {
            int kg = kt * 64 + s * 16 + ln;
            float am = amask[(size_t)b * T_ + kg];
            #pragma unroll
            for (int r = 0; r < 8; ++r) {
                int qg = q0 + wv * 16 + r + hi * 8;
                sc[s][r] = (kg <= qg) ? (sacc[s][r] + am) : -1e30f;
            }
        }

        // online softmax: row max / rescale / row sum (rows live in 16-lane halves)
        float rmax[8], rsum[8];
        #pragma unroll
        for (int r = 0; r < 8; ++r) {
            float v0 = fmaxf(fmaxf(sc[0][r], sc[1][r]), fmaxf(sc[2][r], sc[3][r]));
            #pragma unroll
            for (int d = 1; d < 16; d <<= 1)
                v0 = fmaxf(v0, __shfl_xor(v0, d, 32));
            rmax[r] = v0;
        }
        float corr[8];
        #pragma unroll
        for (int r = 0; r < 8; ++r) {
            float mn = fmaxf(m[r], rmax[r]);
            corr[r] = __expf(m[r] - mn);
            m[r] = mn;
            rsum[r] = 0.f;
        }
        #pragma unroll
        for (int s = 0; s < 4; ++s)
            #pragma unroll
            for (int r = 0; r < 8; ++r) {
                float p = __expf(sc[s][r] - m[r]);
                ps[wv * 16 + r + hi * 8][s * 16 + ln] = f2bf(p);
                rsum[r] += p;
            }
        #pragma unroll
        for (int r = 0; r < 8; ++r) {
            float v0 = rsum[r];
            #pragma unroll
            for (int d = 1; d < 16; d <<= 1)
                v0 += __shfl_xor(v0, d, 32);
            l[r] = l[r] * corr[r] + v0;
        }
        #pragma unroll
        for (int s = 0; s < 4; ++s)
            #pragma unroll
            for (int r = 0; r < 8; ++r)
                o[s][r] *= corr[r];

        // P is consumed by the same wave that wrote it (rows wv*16..wv*16+15);
        // same-wave DS ops are in-order, so no block barrier needed. Pin the
        // instruction scheduler anyway.
        __builtin_amdgcn_sched_barrier(0);

        // O += P V   (contraction over 64 keys = 2 WMMA k-chunks)
        #pragma unroll
        for (int kc = 0; kc < 2; ++kc) {
            Frag16 pa;
            #pragma unroll
            for (int e = 0; e < 16; ++e) {
                int k = kc * 32 + (e >> 3) * 16 + hi * 8 + (e & 7);
                pa.u[e] = ps[wv * 16 + ln][k];
            }
            #pragma unroll
            for (int s = 0; s < 4; ++s) {
                Frag16 vb;   // B[k=key][n=d] = V[key][d]
                #pragma unroll
                for (int e = 0; e < 16; ++e)
                    vb.u[e] = vs[cur][kc * 32 + (e >> 3) * 16 + hi * 8 + (e & 7)][s * 16 + ln];
                o[s] = __builtin_amdgcn_wmma_f32_16x16x32_bf16(
                    false, pa.v, false, vb.v, (short)0, o[s], false, false);
            }
        }
    }

    // normalize and merge heads: out[b, t, h*64 + d]
    #pragma unroll
    for (int r = 0; r < 8; ++r) {
        int t = q0 + wv * 16 + r + hi * 8;
        float inv = 1.0f / l[r];
        #pragma unroll
        for (int s = 0; s < 4; ++s)
            out[((size_t)b * T_ + t) * HID_ + h * 64 + s * 16 + ln] = o[s][r] * inv;
    }
}

// ---------------------------------------------------------------------------
extern "C" void kernel_launch(void* const* d_in, const int* in_sizes, int n_in,
                              void* d_out, int out_size, void* d_ws, size_t ws_size,
                              hipStream_t stream) {
    const float* hs = (const float*)d_in[0];   // [B,T,HID]
    const float* am = (const float*)d_in[1];   // [B,1,1,T]
    const float* Wq = (const float*)d_in[2];
    const float* bq = (const float*)d_in[3];
    const float* Wk = (const float*)d_in[4];
    const float* bk = (const float*)d_in[5];
    const float* Wv = (const float*)d_in[6];
    const float* bv = (const float*)d_in[7];

    const size_t N = (size_t)B_ * H_ * T_ * D_;      // elems per Q/K/V
    unsigned short* q = (unsigned short*)d_ws;       // bf16 [B,H,T,D]
    unsigned short* k = q + N;
    unsigned short* v = k + N;

    dim3 g1(B_ * T_ / 128, H_, 3);
    qkv_gemm_kernel<<<g1, 256, 0, stream>>>(hs, Wq, Wk, Wv, bq, bk, bv, q, k, v);

    dim3 g2(T_ / 64, B_ * H_);
    attn_kernel<<<g2, 128, 0, stream>>>(q, k, v, am, (float*)d_out);
}
